// Mamba_8435315770097
// MI455X (gfx1250) — compile-verified, matrix operand order checked by reading, not run on hardware
//
#include <hip/hip_runtime.h>
#include <hip/hip_bf16.h>

#define L_SEQ   2048
#define DIM     2048
#define NST     16
#define DTR     64
#define KCONV   4
#define DMODEL  1024
#define RTOT    96   // DTR + 2*NST

typedef __attribute__((ext_vector_type(16))) __bf16       v16bf;
typedef __attribute__((ext_vector_type(8)))  float        v8f;
typedef __attribute__((ext_vector_type(4)))  unsigned int v4u;

union Frag { v16bf v; v4u q[2]; };

__device__ __forceinline__ unsigned short f2bf(float f) {
    unsigned int u = __float_as_uint(f);
    u += 0x7FFFu + ((u >> 16) & 1u);   // round-to-nearest-even
    return (unsigned short)(u >> 16);
}

__device__ __forceinline__ void load_frag(Frag& f, const unsigned short* p) {
    f.q[0] = *(const v4u*)(p);
    f.q[1] = *(const v4u*)(p + 16);
}

// ---------------------------------------------------------------------------
// Weight conversion f32 -> bf16 (x_proj, delta_proj, out_proj)
// ---------------------------------------------------------------------------
__global__ void cvt_weights_kernel(const float* __restrict__ xp,
                                   const float* __restrict__ dp,
                                   const float* __restrict__ op,
                                   unsigned short* __restrict__ xpb,
                                   unsigned short* __restrict__ dpb,
                                   unsigned short* __restrict__ opb) {
    const int n1 = RTOT * DIM;
    const int n2 = DIM * DTR;
    const int n3 = DMODEL * DIM;
    const int tot = n1 + n2 + n3;
    for (int i = blockIdx.x * blockDim.x + threadIdx.x; i < tot;
         i += gridDim.x * blockDim.x) {
        if (i < n1)            xpb[i]           = f2bf(xp[i]);
        else if (i < n1 + n2)  dpb[i - n1]      = f2bf(dp[i - n1]);
        else                   opb[i - n1 - n2] = f2bf(op[i - n1 - n2]);
    }
}

// ---------------------------------------------------------------------------
// Depthwise causal conv1d (K=4) + bias + SiLU, time-major outputs [l][d]
// ---------------------------------------------------------------------------
__global__ void conv_silu_kernel(const float* __restrict__ xz,
                                 const float* __restrict__ w,
                                 const float* __restrict__ b,
                                 float* __restrict__ convT,
                                 unsigned short* __restrict__ convT_bf) {
    const int l = blockIdx.x;
    for (int d = threadIdx.x; d < DIM; d += blockDim.x) {
        const float* xr = xz + (size_t)d * L_SEQ;   // x half of xz
        float acc = b[d];
#pragma unroll
        for (int k = 0; k < KCONV; ++k) {
            int li = l + k - (KCONV - 1);
            if (li >= 0) acc += w[d * KCONV + k] * xr[li];
        }
        float s = acc / (1.0f + __expf(-acc));      // silu
        convT[(size_t)l * DIM + d] = s;
        convT_bf[(size_t)l * DIM + d] = f2bf(s);
    }
}

// ---------------------------------------------------------------------------
// NT output tiles (16 M rows x NT*16 N cols) per wave. A fragment shared
// across NT WMMAs per k-step. Software-pipelined: fragments for step k0+32
// are loaded before the WMMAs of step k0 execute, so matrix ops overlap with
// the loads of the next step (no loadcnt-0 stalls). Layouts per ISA tables:
// lane r = row/col index, two 8-element K runs at koff and koff+16.
// ---------------------------------------------------------------------------
template <int NT>
__device__ __forceinline__ void wmma_strip_bf16(const unsigned short* __restrict__ Abf,
                                                int lda,
                                                const unsigned short* __restrict__ Bbf,
                                                int ldb,
                                                int m0, int n0, int kTot, int lane,
                                                v8f c[NT]) {
    const int r    = lane & 15;
    const int koff = (lane >> 4) << 3;   // 0 or 8
    const unsigned short* arow = Abf + (size_t)(m0 + r) * lda + koff;
    const unsigned short* brow = Bbf + (size_t)(n0 + r) * ldb + koff;

    Frag a, bm[NT];
    load_frag(a, arow);
#pragma unroll
    for (int j = 0; j < NT; ++j)
        load_frag(bm[j], brow + (size_t)(j * 16) * ldb);

    for (int k0 = 32; k0 < kTot; k0 += 32) {
        Frag an, bn[NT];
        load_frag(an, arow + k0);
#pragma unroll
        for (int j = 0; j < NT; ++j)
            load_frag(bn[j], brow + (size_t)(j * 16) * ldb + k0);
        // prefetch one further k-step ahead (unconditional, branch-free)
        __builtin_prefetch(arow + k0 + 32, 0, 3);
#pragma unroll
        for (int j = 0; j < NT; ++j)
            c[j] = __builtin_amdgcn_wmma_f32_16x16x32_bf16(
                       false, a.v, false, bm[j].v, (short)0, c[j], false, false);
        a = an;
#pragma unroll
        for (int j = 0; j < NT; ++j)
            bm[j] = bn[j];
    }
#pragma unroll
    for (int j = 0; j < NT; ++j)
        c[j] = __builtin_amdgcn_wmma_f32_16x16x32_bf16(
                   false, a.v, false, bm[j].v, (short)0, c[j], false, false);
}

// ---------------------------------------------------------------------------
// x_dbl = convT (L x DIM) * x_proj^T (DIM x 96); one wave does the full
// 6-tile N row for its 16 L rows. -> [l][96] f32 + bf16
// ---------------------------------------------------------------------------
__global__ void xdbl_gemm_kernel(const unsigned short* __restrict__ convT_bf,
                                 const unsigned short* __restrict__ Wxp_bf,
                                 float* __restrict__ xdbl,
                                 unsigned short* __restrict__ xdbl_bf) {
    const int wave = (blockIdx.x * blockDim.x + threadIdx.x) >> 5;
    const int lane = threadIdx.x & 31;
    const int m0 = wave * 16;                 // waves = L/16 = 128
    v8f c[6] = {};
    wmma_strip_bf16<6>(convT_bf, DIM, Wxp_bf, DIM, m0, 0, DIM, lane, c);
    const int n = lane & 15, rbase = (lane >> 4) * 8;
#pragma unroll
    for (int j = 0; j < 6; ++j) {
#pragma unroll
        for (int v = 0; v < 8; ++v) {
            int l = m0 + rbase + v;
            float val = c[j][v];
            xdbl[(size_t)l * RTOT + j * 16 + n] = val;
            xdbl_bf[(size_t)l * RTOT + j * 16 + n] = f2bf(val);
        }
    }
}

// ---------------------------------------------------------------------------
// deltaT[l][d] = softplus( x_dbl[l][0:64] * delta_proj[d][0:64] + bias[d] )
// 4 N-tiles per wave.
// ---------------------------------------------------------------------------
__global__ void delta_gemm_kernel(const unsigned short* __restrict__ xdbl_bf,
                                  const unsigned short* __restrict__ Wdp_bf,
                                  const float* __restrict__ delta_bias,
                                  float* __restrict__ deltaT) {
    const int wave = (blockIdx.x * blockDim.x + threadIdx.x) >> 5;
    const int lane = threadIdx.x & 31;
    const int nGrp = DIM / 64;                // 32 groups of 4 tiles
    const int m0 = (wave / nGrp) * 16;
    const int n0 = (wave % nGrp) * 64;
    v8f c[4] = {};
    wmma_strip_bf16<4>(xdbl_bf, RTOT, Wdp_bf, DTR, m0, n0, DTR, lane, c);
    const int n = lane & 15, rbase = (lane >> 4) * 8;
#pragma unroll
    for (int j = 0; j < 4; ++j) {
        const int d = n0 + j * 16 + n;
        const float db = delta_bias[d];
#pragma unroll
        for (int v = 0; v < 8; ++v) {
            float x = c[j][v] + db;
            float sp = (x > 20.0f) ? x : log1pf(__expf(x));   // softplus
            deltaT[(size_t)(m0 + rbase + v) * DIM + d] = sp;
        }
    }
}

// ---------------------------------------------------------------------------
// Selective scan. 16 lanes per channel d (one per state n). Per t:
//   h = exp(dt*A)*h + dt*B_t*u ;  y = sum_n h*C_t ; fused +D*u and *silu(z).
// ---------------------------------------------------------------------------
__global__ void scan_kernel(const float* __restrict__ deltaT,
                            const float* __restrict__ convT,
                            const float* __restrict__ xdbl,
                            const float* __restrict__ A,
                            const float* __restrict__ Dv,
                            const float* __restrict__ xz,
                            unsigned short* __restrict__ y_bf) {
    const int tid = threadIdx.x;
    const int n = tid & 15;
    const int d = blockIdx.x * (blockDim.x >> 4) + (tid >> 4);
    const float a  = A[(size_t)d * NST + n];
    const float Dd = Dv[d];
    const float* zrow = xz + (size_t)(DIM + d) * L_SEQ;   // z half of xz
    float h = 0.0f;
    for (int t = 0; t < L_SEQ; ++t) {
        float dt = deltaT[(size_t)t * DIM + d];
        float u  = convT[(size_t)t * DIM + d];
        float Bt = xdbl[(size_t)t * RTOT + DTR + n];
        float Ct = xdbl[(size_t)t * RTOT + DTR + NST + n];
        h = __expf(dt * a) * h + dt * Bt * u;
        float y = h * Ct;
        y += __shfl_xor(y, 1, 32);
        y += __shfl_xor(y, 2, 32);
        y += __shfl_xor(y, 4, 32);
        y += __shfl_xor(y, 8, 32);
        if (n == 0) {
            float z  = zrow[t];
            float sz = z / (1.0f + __expf(-z));
            y_bf[(size_t)t * DIM + d] = f2bf((y + Dd * u) * sz);
        }
    }
}

// ---------------------------------------------------------------------------
// out[l][e] = y (L x DIM) * out_proj^T (DIM x DMODEL) + bias[e]; 4 tiles/wave.
// ---------------------------------------------------------------------------
__global__ void out_gemm_kernel(const unsigned short* __restrict__ y_bf,
                                const unsigned short* __restrict__ Wout_bf,
                                const float* __restrict__ obias,
                                float* __restrict__ out) {
    const int wave = (blockIdx.x * blockDim.x + threadIdx.x) >> 5;
    const int lane = threadIdx.x & 31;
    const int nGrp = DMODEL / 64;             // 16 groups of 4 tiles
    const int m0 = (wave / nGrp) * 16;
    const int n0 = (wave % nGrp) * 64;
    v8f c[4] = {};
    wmma_strip_bf16<4>(y_bf, DIM, Wout_bf, DIM, m0, n0, DIM, lane, c);
    const int n = lane & 15, rbase = (lane >> 4) * 8;
#pragma unroll
    for (int j = 0; j < 4; ++j) {
        const int e = n0 + j * 16 + n;
        const float bb = obias[e];
#pragma unroll
        for (int v = 0; v < 8; ++v)
            out[(size_t)(m0 + rbase + v) * DMODEL + e] = c[j][v] + bb;
    }
}

// ---------------------------------------------------------------------------
extern "C" void kernel_launch(void* const* d_in, const int* in_sizes, int n_in,
                              void* d_out, int out_size, void* d_ws, size_t ws_size,
                              hipStream_t stream) {
    const float* xz     = (const float*)d_in[0];
    const float* convw  = (const float*)d_in[1];
    const float* convb  = (const float*)d_in[2];
    const float* Wxp    = (const float*)d_in[3];
    const float* Wdp    = (const float*)d_in[4];
    const float* Wout   = (const float*)d_in[5];
    const float* obias  = (const float*)d_in[6];
    const float* A      = (const float*)d_in[7];
    const float* Dv     = (const float*)d_in[8];
    const float* dbias  = (const float*)d_in[9];
    float* out = (float*)d_out;

    char* ws = (char*)d_ws;
    size_t off = 0;
    auto take = [&](size_t bytes) -> char* {
        char* p = ws + off;
        off += (bytes + 255) & ~(size_t)255;
        return p;
    };
    float*          convT    = (float*)         take((size_t)L_SEQ * DIM * 4);
    unsigned short* convT_bf = (unsigned short*)take((size_t)L_SEQ * DIM * 2);
    float*          xdbl     = (float*)         take((size_t)L_SEQ * RTOT * 4);
    unsigned short* xdbl_bf  = (unsigned short*)take((size_t)L_SEQ * RTOT * 2);
    float*          deltaT   = (float*)         take((size_t)L_SEQ * DIM * 4);
    unsigned short* y_bf     = (unsigned short*)take((size_t)L_SEQ * DIM * 2);
    unsigned short* Wxp_bf   = (unsigned short*)take((size_t)RTOT * DIM * 2);
    unsigned short* Wdp_bf   = (unsigned short*)take((size_t)DIM * DTR * 2);
    unsigned short* Wout_bf  = (unsigned short*)take((size_t)DMODEL * DIM * 2);

    cvt_weights_kernel<<<1024, 256, 0, stream>>>(Wxp, Wdp, Wout,
                                                 Wxp_bf, Wdp_bf, Wout_bf);
    conv_silu_kernel<<<L_SEQ, 256, 0, stream>>>(xz, convw, convb,
                                                convT, convT_bf);
    // L/16 = 128 waves, full 96-col N row each -> 16 blocks of 8 waves
    xdbl_gemm_kernel<<<16, 256, 0, stream>>>(convT_bf, Wxp_bf, xdbl, xdbl_bf);
    // (L/16)*(DIM/64) = 4096 waves -> 512 blocks
    delta_gemm_kernel<<<512, 256, 0, stream>>>(xdbl_bf, Wdp_bf, dbias, deltaT);
    // DIM*NST = 32768 threads -> 128 blocks of 256
    scan_kernel<<<128, 256, 0, stream>>>(deltaT, convT, xdbl, A, Dv, xz, y_bf);
    // (L/16)*(DMODEL/64) = 2048 waves -> 256 blocks
    out_gemm_kernel<<<256, 256, 0, stream>>>(y_bf, Wout_bf, obias, out);
}
